// Attention_29635274342682
// MI455X (gfx1250) — matrix operand order repaired and checked by reading, not compiled, not run on hardware
//
#include <hip/hip_runtime.h>

typedef __attribute__((ext_vector_type(16))) _Float16 v16h;
typedef __attribute__((ext_vector_type(8)))  _Float16 v8h;
typedef __attribute__((ext_vector_type(8)))  float    v8f;

#define S_TOK   8192
#define HID     3072
#define NHQ     24
#define NHKV    8
#define HD      128
#define NM      2
#define CHUNK   1024
#define QKV_OUT ((NHQ + 2 * NHKV) * HD)   // 5120
#define HALF_S  (S_TOK / NM)              // 4096

// ---------------------------------------------------------------------------
// WMMA fragment loaders (contiguous-K layouts, same order for A and B, f16, wave32)
// A 16x32: lane&15 = M, 16 contiguous K per lane (koff 0 for lanes<16, 16 for >=16)
// B 32x16: lane&15 = N, 16 contiguous K per lane (koff 0 for lanes<16, 16 for >=16)
// C/D f32 16x16: VGPR v -> row v (lanes<16) / v+8 (lanes>=16); col = lane&15
// ---------------------------------------------------------------------------
__device__ __forceinline__ v16h load_a16(const _Float16* __restrict__ p, int ld,
                                         int m0, int k0, int lane) {
    int m  = m0 + (lane & 15);
    int kb = k0 + ((lane & 16) ? 16 : 0);
    const _Float16* q = p + (long)m * ld + kb;
    union { v16h v; v8h h[2]; } u;
    u.h[0] = *(const v8h*)(q);
    u.h[1] = *(const v8h*)(q + 8);
    return u.v;
}

__device__ __forceinline__ v16h load_b16(const _Float16* __restrict__ p, int ld,
                                         int n0, int k0, int lane) {
    int n  = n0 + (lane & 15);
    int kb = k0 + ((lane & 16) ? 16 : 0);
    const _Float16* q = p + (long)n * ld + kb;
    union { v16h v; v8h h[2]; } u;
    u.h[0] = *(const v8h*)(q);
    u.h[1] = *(const v8h*)(q + 8);
    return u.v;
}

// ---------------------------------------------------------------------------
// fp32 -> fp16 conversion (grid-stride)
// ---------------------------------------------------------------------------
__global__ void cvt_f32_f16(const float* __restrict__ src, _Float16* __restrict__ dst, int n) {
    int i = blockIdx.x * blockDim.x + threadIdx.x;
    int stride = gridDim.x * blockDim.x;
    for (; i < n; i += stride) dst[i] = (_Float16)src[i];
}

// ---------------------------------------------------------------------------
// MoE GEMM: C[r, c] = sum_k A[r,k] * Bw[m, c, k] + bias[m, c],  m = r / 4096
// Block tile 128x256 (8 waves, 2M x 4N), wave tile 64x64 (4x4 WMMA tiles).
// 16 WMMAs per 8 fragment loads per k-step -> ~32 flop/byte from cache.
// ---------------------------------------------------------------------------
__global__ __launch_bounds__(256) void gemm_moe(const _Float16* __restrict__ A,
                                                const _Float16* __restrict__ Bw,
                                                const float* __restrict__ bias,
                                                float* __restrict__ C,
                                                int Kd, int Nd) {
    int lane = threadIdx.x & 31;
    int wv   = threadIdx.x >> 5;
    int row0 = blockIdx.x * 128 + (wv & 1) * 64;
    int col0 = blockIdx.y * 256 + (wv >> 1) * 64;
    int m    = (blockIdx.x * 128) / HALF_S;           // 128 | 4096 -> no straddle
    const _Float16* Bm = Bw + (long)m * Nd * Kd;
    const float*    bm = bias + (long)m * Nd;

    v8f acc[4][4] = {};
    for (int k0 = 0; k0 < Kd; k0 += 32) {
        v16h a[4], b[4];
#pragma unroll
        for (int i = 0; i < 4; ++i) a[i] = load_a16(A, Kd, row0 + i * 16, k0, lane);
#pragma unroll
        for (int j = 0; j < 4; ++j) b[j] = load_b16(Bm, Kd, col0 + j * 16, k0, lane);
#pragma unroll
        for (int i = 0; i < 4; ++i)
#pragma unroll
            for (int j = 0; j < 4; ++j)
                acc[i][j] = __builtin_amdgcn_wmma_f32_16x16x32_f16(
                    false, a[i], false, b[j], (short)0, acc[i][j], false, false);
    }
#pragma unroll
    for (int j = 0; j < 4; ++j) {
        int col  = col0 + j * 16 + (lane & 15);
        float bv = bm[col];
#pragma unroll
        for (int i = 0; i < 4; ++i) {
#pragma unroll
            for (int v = 0; v < 8; ++v) {
                int r = row0 + i * 16 + v + ((lane & 16) ? 8 : 0);
                C[(long)r * Nd + col] = acc[i][j][v] + bv;
            }
        }
    }
}

// ---------------------------------------------------------------------------
// QK RMSNorm + RoPE + V transpose.  One wave per (token, head-slot).
// slots 0..23 = q heads (norm+rope, scaled by 1/sqrt(HD)), 24..31 = k heads
// (norm+rope), 32..39 = v heads (convert + transpose to [kv*HD + d][token]).
// Lane l owns elements {2l, 2l+1, 2l+64, 2l+65} so RoPE needs no cross-lane.
// ---------------------------------------------------------------------------
__global__ __launch_bounds__(256) void qk_norm_rope(const float* __restrict__ qkv,
                                                    const float* __restrict__ qw,
                                                    const float* __restrict__ kw,
                                                    const float* __restrict__ cosT,
                                                    const float* __restrict__ sinT,
                                                    _Float16* __restrict__ qh,
                                                    _Float16* __restrict__ kh,
                                                    _Float16* __restrict__ vT) {
    int lane = threadIdx.x & 31;
    int gw   = blockIdx.x * (blockDim.x >> 5) + (threadIdx.x >> 5);
    int t    = gw / 40;
    int hi   = gw % 40;
    if (t >= S_TOK) return;
    int j0 = 2 * lane;

    if (hi >= 32) {                                   // V: convert + transpose
        int kv = hi - 32;
        const float* src = qkv + (long)t * QKV_OUT + (NHQ + NHKV) * HD + kv * HD;
        vT[(long)(kv * HD + j0)      * S_TOK + t] = (_Float16)src[j0];
        vT[(long)(kv * HD + j0 + 1)  * S_TOK + t] = (_Float16)src[j0 + 1];
        vT[(long)(kv * HD + j0 + 64) * S_TOK + t] = (_Float16)src[j0 + 64];
        vT[(long)(kv * HD + j0 + 65) * S_TOK + t] = (_Float16)src[j0 + 65];
        return;
    }

    int m = t / HALF_S;
    const float* src;
    const float* nw;
    _Float16* dst;
    float outscale;
    if (hi < NHQ) {
        src = qkv + (long)t * QKV_OUT + hi * HD;
        nw  = qw + m * HD;
        dst = qh + ((long)t * NHQ + hi) * HD;
        outscale = 0.08838834764831845f;              // HD^-0.5 folded into q
    } else {
        int kv = hi - NHQ;
        src = qkv + (long)t * QKV_OUT + NHQ * HD + kv * HD;
        nw  = kw + m * HD;
        dst = kh + ((long)t * NHKV + kv) * HD;
        outscale = 1.0f;
    }
    float a0 = src[j0], a1 = src[j0 + 1], b0 = src[j0 + 64], b1 = src[j0 + 65];
    float ss = a0 * a0 + a1 * a1 + b0 * b0 + b1 * b1;
#pragma unroll
    for (int msk = 16; msk >= 1; msk >>= 1) ss += __shfl_xor(ss, msk, 32);
    float r = rsqrtf(ss * (1.0f / HD) + 1e-6f);
    a0 *= r * (nw[j0] + 1.0f);
    a1 *= r * (nw[j0 + 1] + 1.0f);
    b0 *= r * (nw[j0 + 64] + 1.0f);
    b1 *= r * (nw[j0 + 65] + 1.0f);
    float c0 = cosT[(long)t * 64 + j0],     s0 = sinT[(long)t * 64 + j0];
    float c1 = cosT[(long)t * 64 + j0 + 1], s1 = sinT[(long)t * 64 + j0 + 1];
    dst[j0]      = (_Float16)((a0 * c0 - b0 * s0) * outscale);
    dst[j0 + 1]  = (_Float16)((a1 * c1 - b1 * s1) * outscale);
    dst[j0 + 64] = (_Float16)((b0 * c0 + a0 * s0) * outscale);
    dst[j0 + 65] = (_Float16)((b1 * c1 + a1 * s1) * outscale);
}

// ---------------------------------------------------------------------------
// Block-diagonal flash attention.  One wave per (head, 16-row q tile).
// Online softmax; P transposed C-layout -> A-layout through 1KB LDS per wave.
// ---------------------------------------------------------------------------
__global__ __launch_bounds__(256) void attn_block(const _Float16* __restrict__ qh,
                                                  const _Float16* __restrict__ kh,
                                                  const _Float16* __restrict__ vT,
                                                  _Float16* __restrict__ oh) {
    __shared__ _Float16 plds[8][16][32];
    int lane = threadIdx.x & 31;
    int wv   = threadIdx.x >> 5;
    int gw   = blockIdx.x * 8 + wv;
    int h    = gw % NHQ;
    int mt   = gw / NHQ;                               // 0..511
    int row0 = mt * 16;
    int kstart = (row0 / CHUNK) * CHUNK;
    int hk   = h / (NHQ / NHKV);

    v16h qf[4];
#pragma unroll
    for (int kk = 0; kk < 4; ++kk)
        qf[kk] = load_a16(qh + h * HD, NHQ * HD, row0, kk * 32, lane);

    v8f o[8] = {};
    float mrun[8], lrun[8];
#pragma unroll
    for (int v = 0; v < 8; ++v) { mrun[v] = -1e30f; lrun[v] = 0.0f; }

    const _Float16* myp = &plds[wv][0][0];
    int prow = (lane & 16) ? 8 : 0;
    int pcol = lane & 15;

    for (int n0 = 0; n0 < CHUNK; n0 += 32) {
        v8f s0 = {}, s1 = {};
#pragma unroll
        for (int kk = 0; kk < 4; ++kk) {
            v16h b0 = load_b16(kh + hk * HD, NHKV * HD, kstart + n0,      kk * 32, lane);
            v16h b1 = load_b16(kh + hk * HD, NHKV * HD, kstart + n0 + 16, kk * 32, lane);
            s0 = __builtin_amdgcn_wmma_f32_16x16x32_f16(false, qf[kk], false, b0, (short)0, s0, false, false);
            s1 = __builtin_amdgcn_wmma_f32_16x16x32_f16(false, qf[kk], false, b1, (short)0, s1, false, false);
        }
        float corr[8];
#pragma unroll
        for (int v = 0; v < 8; ++v) {
            float mx = fmaxf(s0[v], s1[v]);
#pragma unroll
            for (int msk = 8; msk >= 1; msk >>= 1) mx = fmaxf(mx, __shfl_xor(mx, msk, 32));
            float mnew = fmaxf(mrun[v], mx);
            float c  = __expf(mrun[v] - mnew);
            float p0 = __expf(s0[v] - mnew);
            float p1 = __expf(s1[v] - mnew);
            float ps = p0 + p1;
#pragma unroll
            for (int msk = 8; msk >= 1; msk >>= 1) ps += __shfl_xor(ps, msk, 32);
            lrun[v] = lrun[v] * c + ps;
            mrun[v] = mnew;
            corr[v] = c;
            plds[wv][prow + v][pcol]      = (_Float16)p0;   // C-layout -> LDS
            plds[wv][prow + v][16 + pcol] = (_Float16)p1;
        }
#pragma unroll
        for (int j = 0; j < 8; ++j)
#pragma unroll
            for (int v = 0; v < 8; ++v) o[j][v] *= corr[v];

        v16h pa = load_a16(myp, 32, 0, 0, lane);            // A-layout from LDS
#pragma unroll
        for (int j = 0; j < 8; ++j) {
            v16h vb = load_b16(vT + (long)hk * HD * S_TOK, S_TOK, j * 16, kstart + n0, lane);
            o[j] = __builtin_amdgcn_wmma_f32_16x16x32_f16(false, pa, false, vb, (short)0, o[j], false, false);
        }
    }
#pragma unroll
    for (int v = 0; v < 8; ++v) lrun[v] = 1.0f / lrun[v];
    int rbase = row0 + ((lane & 16) ? 8 : 0);
    int cbase = h * HD + (lane & 15);
#pragma unroll
    for (int j = 0; j < 8; ++j)
#pragma unroll
        for (int v = 0; v < 8; ++v)
            oh[(long)(rbase + v) * HID + cbase + j * 16] = (_Float16)(o[j][v] * lrun[v]);
}

// ---------------------------------------------------------------------------
extern "C" void kernel_launch(void* const* d_in, const int* in_sizes, int n_in,
                              void* d_out, int out_size, void* d_ws, size_t ws_size,
                              hipStream_t stream) {
    const float* x      = (const float*)d_in[0];
    const float* w_qkv  = (const float*)d_in[1];
    const float* b_qkv  = (const float*)d_in[2];
    const float* q_nw   = (const float*)d_in[3];
    const float* k_nw   = (const float*)d_in[4];
    const float* cosT   = (const float*)d_in[5];
    const float* sinT   = (const float*)d_in[6];
    const float* w_out  = (const float*)d_in[7];
    const float* b_out  = (const float*)d_in[8];
    float* out = (float*)d_out;

    char* ws = (char*)d_ws;
    size_t off = 0;
    auto carve = [&](size_t bytes) { char* p = ws + off; off += (bytes + 255) & ~(size_t)255; return p; };

    _Float16* xh    = (_Float16*)carve((size_t)S_TOK * HID * 2);
    _Float16* wqkvh = (_Float16*)carve((size_t)NM * QKV_OUT * HID * 2);
    _Float16* wouth = (_Float16*)carve((size_t)NM * HID * HID * 2);
    float*    qkv   = (float*)   carve((size_t)S_TOK * QKV_OUT * 4);
    _Float16* qh    = (_Float16*)carve((size_t)S_TOK * NHQ * HD * 2);
    _Float16* khb   = (_Float16*)carve((size_t)S_TOK * NHKV * HD * 2);
    _Float16* vT    = (_Float16*)carve((size_t)S_TOK * NHKV * HD * 2);
    _Float16* oh    = (_Float16*)carve((size_t)S_TOK * NHQ * HD * 2);

    cvt_f32_f16<<<8192, 256, 0, stream>>>(x, xh, S_TOK * HID);
    cvt_f32_f16<<<8192, 256, 0, stream>>>(w_qkv, wqkvh, NM * QKV_OUT * HID);
    cvt_f32_f16<<<8192, 256, 0, stream>>>(w_out, wouth, NM * HID * HID);

    gemm_moe<<<dim3(S_TOK / 128, QKV_OUT / 256), 256, 0, stream>>>(xh, wqkvh, b_qkv, qkv, HID, QKV_OUT);

    qk_norm_rope<<<(S_TOK * 40) / 8, 256, 0, stream>>>(qkv, q_nw, k_nw, cosT, sinT, qh, khb, vT);

    attn_block<<<(NHQ * (S_TOK / 16)) / 8, 256, 0, stream>>>(qh, khb, vT, oh);

    gemm_moe<<<dim3(S_TOK / 128, HID / 256), 256, 0, stream>>>(oh, wouth, b_out, out, HID, HID);
}
